// TransformerDecoderLayerWithCache_84963043049666
// MI455X (gfx1250) — matrix operand order repaired and checked by reading, not compiled, not run on hardware
//
#include <hip/hip_runtime.h>
#include <hip/hip_bf16.h>

// ---------------------------------------------------------------------------
// MI455X (gfx1250, wave32) transformer decoder layer.
// All matmuls go through v_wmma_f32_16x16x32_bf16 (f32 accumulate).
// ---------------------------------------------------------------------------

typedef __attribute__((ext_vector_type(16))) __bf16 v16bf;
typedef __attribute__((ext_vector_type(8)))  __bf16 v8bf;
typedef __attribute__((ext_vector_type(8)))  float  v8f;

#define D_MODEL 1024
#define NHEAD   16
#define HEADD   64
#define SEQ     1024
#define ROWS    4096      // B * S
#define DFF_    4096
#define NB      4         // batch

__device__ __forceinline__ v8f vzero8() {
  v8f z;
#pragma unroll
  for (int i = 0; i < 8; ++i) z[i] = 0.0f;
  return z;
}

__device__ __forceinline__ v8f wmma_bf16(v16bf a, v16bf b, v8f c) {
  // D = A(16x32) * B(32x16) + C, f32 accumulate
  return __builtin_amdgcn_wmma_f32_16x16x32_bf16(false, a, false, b, (short)0, c,
                                                 false, false);
}

// A fragment: 16x32 (MxK). lanes 0-15: rows, K 0..7 & 16..23; lanes 16-31: K 8..15 & 24..31.
__device__ __forceinline__ v16bf load_a_frag(const __bf16* base, int ld) {
  const int lane = threadIdx.x & 31;
  const int m  = lane & 15;
  const int ks = lane >> 4;
  const __bf16* p = base + m * ld + ks * 8;
  v8bf lo = *(const v8bf*)p;          // K = ks*8 .. ks*8+7        -> halves 0..7
  v8bf hi = *(const v8bf*)(p + 16);   // K = 16+ks*8 .. 16+ks*8+7  -> halves 8..15
  v16bf r;
#pragma unroll
  for (int i = 0; i < 8; ++i) { r[i] = lo[i]; r[i + 8] = hi[i]; }
  return r;
}

// B fragment: 32x16 (KxN) from row-major [N,K] storage (W[n,k] / K-cache rows).
// lane n = lane&15; lanes 0-15 hold K 0..7 & 16..23, lanes 16-31 hold K 8..15 & 24..31.
__device__ __forceinline__ v16bf load_b_frag(const __bf16* base, int ld) {
  const int lane = threadIdx.x & 31;
  const int n  = lane & 15;
  const int ks = lane >> 4;
  const __bf16* p = base + n * ld + ks * 8;
  v8bf lo = *(const v8bf*)p;
  v8bf hi = *(const v8bf*)(p + 16);
  v16bf r;
#pragma unroll
  for (int i = 0; i < 8; ++i) { r[i] = lo[i]; r[i + 8] = hi[i]; }
  return r;
}

// ---------------------------------------------------------------------------
// fp32 -> bf16 convert
// ---------------------------------------------------------------------------
__global__ void k_cvt(const float* __restrict__ in, __bf16* __restrict__ out, int n) {
  int i = blockIdx.x * blockDim.x + threadIdx.x;
  const int stride = gridDim.x * blockDim.x;
  for (; i < n; i += stride) out[i] = (__bf16)in[i];
}

// ---------------------------------------------------------------------------
// LayerNorm over D=1024, bf16 output. One block (256 thr) per row.
// ---------------------------------------------------------------------------
__global__ __launch_bounds__(256) void k_ln(const float* __restrict__ x,
                                            const float* __restrict__ g,
                                            const float* __restrict__ beta,
                                            __bf16* __restrict__ out) {
  const int row  = blockIdx.x;
  const int tid  = threadIdx.x;
  const int lane = tid & 31;
  const int w    = tid >> 5;
  const float* xr = x + (long)row * D_MODEL;
  float v[4];
  float s = 0.f, ss = 0.f;
#pragma unroll
  for (int i = 0; i < 4; ++i) {
    v[i] = xr[tid + i * 256];
    s += v[i];
    ss += v[i] * v[i];
  }
#pragma unroll
  for (int m = 1; m < 32; m <<= 1) {
    s  += __shfl_xor(s, m, 32);
    ss += __shfl_xor(ss, m, 32);
  }
  __shared__ float sb[8], ssb[8];
  if (lane == 0) { sb[w] = s; ssb[w] = ss; }
  __syncthreads();
  float st = 0.f, sst = 0.f;
#pragma unroll
  for (int i = 0; i < 8; ++i) { st += sb[i]; sst += ssb[i]; }
  const float mu  = st * (1.0f / D_MODEL);
  const float var = sst * (1.0f / D_MODEL) - mu * mu;
  const float rs  = rsqrtf(var + 1e-5f);
  __bf16* orow = out + (long)row * D_MODEL;
#pragma unroll
  for (int i = 0; i < 4; ++i) {
    const int c = tid + i * 256;
    orow[c] = (__bf16)((v[i] - mu) * rs * g[c] + beta[c]);
  }
}

// ---------------------------------------------------------------------------
// GEMM: C[M,N] = act(A[M,K] @ W[N,K]^T + bias) (+resid). A,W bf16; C fp32/bf16.
// heads=1 remaps (m,n) -> [B,H,S,64] layout for attention operands & k/v cache.
// Block: 256 thr = 8 waves, tile 128x128, K-step 32. LDS pad stride 40 elems.
// ---------------------------------------------------------------------------
#define TM 128
#define TN 128
#define TK 32
#define APAD 40

__global__ __launch_bounds__(256) void k_gemm(
    const __bf16* __restrict__ A, const __bf16* __restrict__ W,
    const float* __restrict__ bias, const float* __restrict__ resid,
    float* __restrict__ outF, __bf16* __restrict__ outB,
    int K, int N, int relu, int heads) {
  __shared__ __align__(16) __bf16 As[TM * APAD];
  __shared__ __align__(16) __bf16 Bs[TN * APAD];
  const int tid  = threadIdx.x;
  const int lane = tid & 31;
  const int w    = tid >> 5;
  const int wm   = w >> 2;   // 0..1  (64 rows each)
  const int wn   = w & 3;    // 0..3  (32 cols each)
  const long m0 = (long)blockIdx.y * TM;
  const long n0 = (long)blockIdx.x * TN;

  v8f acc[4][2];
#pragma unroll
  for (int i = 0; i < 4; ++i)
#pragma unroll
    for (int j = 0; j < 2; ++j) acc[i][j] = vzero8();

  const int row = tid >> 1;
  const int seg = (tid & 1) * 16;
  for (int k0 = 0; k0 < K; k0 += TK) {
    __syncthreads();
    {
      const __bf16* ga = A + (m0 + row) * (long)K + k0 + seg;
      *(v8bf*)(As + row * APAD + seg)     = *(const v8bf*)ga;
      *(v8bf*)(As + row * APAD + seg + 8) = *(const v8bf*)(ga + 8);
      const __bf16* gw = W + (n0 + row) * (long)K + k0 + seg;
      *(v8bf*)(Bs + row * APAD + seg)     = *(const v8bf*)gw;
      *(v8bf*)(Bs + row * APAD + seg + 8) = *(const v8bf*)(gw + 8);
    }
    __syncthreads();
    v16bf bfrag[2];
#pragma unroll
    for (int nt = 0; nt < 2; ++nt)
      bfrag[nt] = load_b_frag(Bs + (wn * 32 + nt * 16) * APAD, APAD);
#pragma unroll
    for (int mt = 0; mt < 4; ++mt) {
      v16bf afrag = load_a_frag(As + (wm * 64 + mt * 16) * APAD, APAD);
#pragma unroll
      for (int nt = 0; nt < 2; ++nt)
        acc[mt][nt] = wmma_bf16(afrag, bfrag[nt], acc[mt][nt]);
    }
  }

  // C layout: lane n = lane&15; row m = v + (lane>>4)*8
  const int hh = lane >> 4;
  const int nn = lane & 15;
#pragma unroll
  for (int mt = 0; mt < 4; ++mt) {
#pragma unroll
    for (int nt = 0; nt < 2; ++nt) {
      const long nG = n0 + wn * 32 + nt * 16 + nn;
      const float bval = bias ? bias[nG] : 0.0f;
#pragma unroll
      for (int v = 0; v < 8; ++v) {
        const long mG = m0 + wm * 64 + mt * 16 + v + hh * 8;
        float val = acc[mt][nt][v] + bval;
        if (relu) val = fmaxf(val, 0.0f);
        long oi;
        if (!heads) {
          oi = mG * (long)N + nG;
          if (resid) val += resid[oi];
        } else {
          const long bb = mG >> 10, sL = mG & 1023;
          const long h  = nG >> 6,  hd = nG & 63;
          oi = ((bb * NHEAD + h) * SEQ + sL) * HEADD + hd;
        }
        if (outF) outF[oi] = val;
        if (outB) outB[oi] = (__bf16)val;
      }
    }
  }
}

// ---------------------------------------------------------------------------
// Flash attention. Block: 128 queries x (b,h); 8 waves own 16 q-rows each.
// KV staged per 64-key block: K row-major, V transposed into LDS.
// Scores & PV both via WMMA bf16; online softmax via xor-butterfly.
// ---------------------------------------------------------------------------
#define KB 64
#define KPAD 72

__global__ __launch_bounds__(256) void k_attn(
    const __bf16* __restrict__ Qh, const __bf16* __restrict__ Kh,
    const __bf16* __restrict__ Vh, __bf16* __restrict__ outB,
    int Skv, int causal) {
  __shared__ __align__(16) __bf16 Ks[KB * KPAD];      // [key][d]
  __shared__ __align__(16) __bf16 Vt[HEADD * KPAD];   // [d][key]
  __shared__ __align__(16) __bf16 Ps[8][16 * KPAD];   // per-wave P tile

  const int tid  = threadIdx.x;
  const int lane = tid & 31;
  const int w    = tid >> 5;
  const int hh   = lane >> 4;
  const int nn   = lane & 15;
  const int b    = blockIdx.z;
  const int h    = blockIdx.y;
  const int qblk = blockIdx.x * 128;
  const int q0   = qblk + w * 16;

  const long headQ  = ((long)(b * NHEAD + h)) * SEQ * HEADD;
  const long headKV = ((long)(b * NHEAD + h)) * (long)Skv * HEADD;
  const __bf16* Qp  = Qh + headQ + (long)q0 * HEADD;

  v16bf qa[2];
  qa[0] = load_a_frag(Qp, HEADD);
  qa[1] = load_a_frag(Qp + 32, HEADD);

  v8f o[4];
  float mrow[8], lsum[8];
#pragma unroll
  for (int dt = 0; dt < 4; ++dt) o[dt] = vzero8();
#pragma unroll
  for (int v = 0; v < 8; ++v) { mrow[v] = -1e30f; lsum[v] = 0.0f; }

  const int kend = causal ? ((qblk + 128 < Skv) ? qblk + 128 : Skv) : Skv;
  const int srow = tid >> 2;
  const int sseg = (tid & 3) * 16;
  for (int kb = 0; kb < kend; kb += KB) {
    __syncthreads();
    {
      const __bf16* kg = Kh + headKV + (long)(kb + srow) * HEADD + sseg;
      *(v8bf*)(Ks + srow * KPAD + sseg)     = *(const v8bf*)kg;
      *(v8bf*)(Ks + srow * KPAD + sseg + 8) = *(const v8bf*)(kg + 8);
      const __bf16* vg = Vh + headKV + (long)(kb + srow) * HEADD + sseg;
#pragma unroll
      for (int j = 0; j < 16; ++j) Vt[(sseg + j) * KPAD + srow] = vg[j];
    }
    __syncthreads();

    // scores: S[16q x 64k] = Q(16x64) * K^T
    v8f sc[4];
#pragma unroll
    for (int nt = 0; nt < 4; ++nt) {
      v8f c = vzero8();
      c = wmma_bf16(qa[0], load_b_frag(Ks + nt * 16 * KPAD, KPAD), c);
      c = wmma_bf16(qa[1], load_b_frag(Ks + nt * 16 * KPAD + 32, KPAD), c);
      sc[nt] = c;
    }

    // scale + mask + per-row max (rows live as (v, lane-half); reduce over 16 lanes)
    float pm[8];
#pragma unroll
    for (int v = 0; v < 8; ++v) pm[v] = -1e30f;
#pragma unroll
    for (int nt = 0; nt < 4; ++nt) {
      const int kj = kb + nt * 16 + nn;
#pragma unroll
      for (int v = 0; v < 8; ++v) {
        float s = sc[nt][v] * 0.125f;  // 1/sqrt(64)
        if (causal && kj > q0 + v + hh * 8) s = -1e30f;
        sc[nt][v] = s;
        pm[v] = fmaxf(pm[v], s);
      }
    }
#pragma unroll
    for (int m = 1; m < 16; m <<= 1)
#pragma unroll
      for (int v = 0; v < 8; ++v) pm[v] = fmaxf(pm[v], __shfl_xor(pm[v], m, 32));

    float corr[8], ls[8];
#pragma unroll
    for (int v = 0; v < 8; ++v) {
      const float mnew = fmaxf(mrow[v], pm[v]);
      corr[v] = __expf(mrow[v] - mnew);
      mrow[v] = mnew;
      ls[v] = 0.0f;
    }
#pragma unroll
    for (int nt = 0; nt < 4; ++nt)
#pragma unroll
      for (int v = 0; v < 8; ++v) {
        const float e = __expf(sc[nt][v] - mrow[v]);
        sc[nt][v] = e;
        ls[v] += e;
      }
#pragma unroll
    for (int m = 1; m < 16; m <<= 1)
#pragma unroll
      for (int v = 0; v < 8; ++v) ls[v] += __shfl_xor(ls[v], m, 32);
#pragma unroll
    for (int v = 0; v < 8; ++v) lsum[v] = lsum[v] * corr[v] + ls[v];
#pragma unroll
    for (int dt = 0; dt < 4; ++dt)
#pragma unroll
      for (int v = 0; v < 8; ++v) o[dt][v] *= corr[v];

    // spill P (C layout) -> per-wave LDS tile; re-read as A fragments (wave-local,
    // LDS ops from the same wave are in-order, no barrier needed).
#pragma unroll
    for (int nt = 0; nt < 4; ++nt)
#pragma unroll
      for (int v = 0; v < 8; ++v)
        Ps[w][(v + hh * 8) * KPAD + nt * 16 + nn] = (__bf16)sc[nt][v];

    v16bf pa0 = load_a_frag(&Ps[w][0], KPAD);
    v16bf pa1 = load_a_frag(&Ps[w][32], KPAD);
#pragma unroll
    for (int dt = 0; dt < 4; ++dt) {
      o[dt] = wmma_bf16(pa0, load_b_frag(Vt + dt * 16 * KPAD, KPAD), o[dt]);
      o[dt] = wmma_bf16(pa1, load_b_frag(Vt + dt * 16 * KPAD + 32, KPAD), o[dt]);
    }
  }

#pragma unroll
  for (int v = 0; v < 8; ++v) lsum[v] = 1.0f / lsum[v];
  const long orow0 = (long)b * SEQ + q0;
#pragma unroll
  for (int dt = 0; dt < 4; ++dt) {
    const int col = h * HEADD + dt * 16 + nn;
#pragma unroll
    for (int v = 0; v < 8; ++v) {
      const long mG = orow0 + v + hh * 8;
      outB[mG * D_MODEL + col] = (__bf16)(o[dt][v] * lsum[v]);
    }
  }
}

// ---------------------------------------------------------------------------
// Host orchestration
// ---------------------------------------------------------------------------
extern "C" void kernel_launch(void* const* d_in, const int* in_sizes, int n_in,
                              void* d_out, int out_size, void* d_ws, size_t ws_size,
                              hipStream_t stream) {
  (void)in_sizes; (void)n_in; (void)out_size; (void)ws_size;

  const float* tgt       = (const float*)d_in[0];
  const float* memory    = (const float*)d_in[1];
  const float* Wq        = (const float*)d_in[2];
  const float* Wk        = (const float*)d_in[3];
  const float* Wv        = (const float*)d_in[4];
  const float* Wo        = (const float*)d_in[5];
  const float* mha_in_w  = (const float*)d_in[6];
  const float* mha_in_b  = (const float*)d_in[7];
  const float* mha_out_w = (const float*)d_in[8];
  const float* mha_out_b = (const float*)d_in[9];
  const float* W1        = (const float*)d_in[10];
  const float* b1        = (const float*)d_in[11];
  const float* W2        = (const float*)d_in[12];
  const float* b2        = (const float*)d_in[13];
  const float* g1  = (const float*)d_in[14];
  const float* be1 = (const float*)d_in[15];
  const float* g2  = (const float*)d_in[16];
  const float* be2 = (const float*)d_in[17];
  const float* g3  = (const float*)d_in[18];
  const float* be3 = (const float*)d_in[19];

  float* out_x = (float*)d_out;                       // [B,S,D]
  float* out_k = out_x + (long)ROWS * D_MODEL;        // [B,H,S,64]
  float* out_v = out_k + (long)ROWS * D_MODEL;        // [B,H,S,64]

  char* base = (char*)d_ws;
  size_t off = 0;
  auto carve = [&](size_t bytes) -> void* {
    void* p = base + off;
    off = (off + bytes + 255) & ~(size_t)255;
    return p;
  };
  const size_t DB = (size_t)D_MODEL * D_MODEL;  // 1M elems

  __bf16* wq_b   = (__bf16*)carve(DB * 2);
  __bf16* wk_b   = (__bf16*)carve(DB * 2);
  __bf16* wv_b   = (__bf16*)carve(DB * 2);
  __bf16* wo_b   = (__bf16*)carve(DB * 2);
  __bf16* win_b  = (__bf16*)carve(3 * DB * 2);        // Wq' | Wk' | Wv'
  __bf16* wout_b = (__bf16*)carve(DB * 2);
  __bf16* w1_b   = (__bf16*)carve((size_t)DFF_ * D_MODEL * 2);
  __bf16* w2_b   = (__bf16*)carve((size_t)DFF_ * D_MODEL * 2);
  __bf16* ln_b   = (__bf16*)carve((size_t)ROWS * D_MODEL * 2);
  __bf16* mem_b  = (__bf16*)carve((size_t)ROWS * D_MODEL * 2);
  __bf16* qh_b   = (__bf16*)carve((size_t)ROWS * D_MODEL * 2);
  __bf16* kh_b   = (__bf16*)carve((size_t)ROWS * D_MODEL * 2);
  __bf16* vh_b   = (__bf16*)carve((size_t)ROWS * D_MODEL * 2);
  __bf16* ao_b   = (__bf16*)carve((size_t)ROWS * D_MODEL * 2);
  __bf16* h1_b   = (__bf16*)carve((size_t)ROWS * DFF_ * 2);
  float*  x1_f   = (float*)carve((size_t)ROWS * D_MODEL * 4);
  float*  x2_f   = (float*)carve((size_t)ROWS * D_MODEL * 4);

  auto cvt = [&](const float* src, __bf16* dst, long n) {
    k_cvt<<<2048, 256, 0, stream>>>(src, dst, (int)n);
  };
  auto gemm = [&](const __bf16* A, const __bf16* W, const float* bias,
                  const float* resid, float* oF, __bf16* oB,
                  int K, int N, int relu, int heads) {
    dim3 grid(N / TN, ROWS / TM);
    k_gemm<<<grid, 256, 0, stream>>>(A, W, bias, resid, oF, oB, K, N, relu, heads);
  };

  // weight / memory conversion
  cvt(Wq, wq_b, (long)DB);
  cvt(Wk, wk_b, (long)DB);
  cvt(Wv, wv_b, (long)DB);
  cvt(Wo, wo_b, (long)DB);
  cvt(mha_in_w, win_b, (long)(3 * DB));
  cvt(mha_out_w, wout_b, (long)DB);
  cvt(W1, w1_b, (long)DFF_ * D_MODEL);
  cvt(W2, w2_b, (long)DFF_ * D_MODEL);
  cvt(memory, mem_b, (long)ROWS * D_MODEL);

  // --- self attention block ---
  k_ln<<<ROWS, 256, 0, stream>>>(tgt, g1, be1, ln_b);
  gemm(ln_b, wq_b, nullptr, nullptr, nullptr, qh_b, D_MODEL, D_MODEL, 0, 1);
  gemm(ln_b, wk_b, nullptr, nullptr, out_k,   kh_b, D_MODEL, D_MODEL, 0, 1);  // sa_k
  gemm(ln_b, wv_b, nullptr, nullptr, out_v,   vh_b, D_MODEL, D_MODEL, 0, 1);  // sa_v
  k_attn<<<dim3(SEQ / 128, NHEAD, NB), 256, 0, stream>>>(qh_b, kh_b, vh_b, ao_b,
                                                         SEQ, 1);
  gemm(ao_b, wo_b, nullptr, tgt, x1_f, nullptr, D_MODEL, D_MODEL, 0, 0);  // x1 = tgt + sa

  // --- cross attention block ---
  k_ln<<<ROWS, 256, 0, stream>>>(x1_f, g2, be2, ln_b);
  gemm(ln_b,  win_b,          mha_in_b,        nullptr, nullptr, qh_b, D_MODEL, D_MODEL, 0, 1);
  gemm(mem_b, win_b + DB,     mha_in_b + 1024, nullptr, nullptr, kh_b, D_MODEL, D_MODEL, 0, 1);
  gemm(mem_b, win_b + 2 * DB, mha_in_b + 2048, nullptr, nullptr, vh_b, D_MODEL, D_MODEL, 0, 1);
  k_attn<<<dim3(SEQ / 128, NHEAD, NB), 256, 0, stream>>>(qh_b, kh_b, vh_b, ao_b,
                                                         SEQ, 0);
  gemm(ao_b, wout_b, mha_out_b, x1_f, x2_f, nullptr, D_MODEL, D_MODEL, 0, 0);  // x2 = x1 + ca

  // --- FFN block ---
  k_ln<<<ROWS, 256, 0, stream>>>(x2_f, g3, be3, ln_b);
  gemm(ln_b, w1_b, b1, nullptr, nullptr, h1_b, D_MODEL, DFF_, 1, 0);           // relu
  gemm(h1_b, w2_b, b2, x2_f, out_x, nullptr, DFF_, D_MODEL, 0, 0);             // x = x2 + ff
}
